// TorchTransformerDecoder_73933567034152
// MI455X (gfx1250) — matrix operand order repaired and checked by reading, not compiled, not run on hardware
//
#include <hip/hip_runtime.h>
#include <math.h>

#define TI_ 128
#define NB_ 8
#define TO_ 32
#define TT_ 33
#define D_  512
#define NH_ 8
#define HD_ 64
#define FF_ 2048
#define VV_ 2000
#define LL_ 6
#define LDA_ 520   // 512 + 8 halves padding -> breaks LDS bank conflicts

typedef __attribute__((ext_vector_type(16))) _Float16 v16h;
typedef __attribute__((ext_vector_type(8)))  float    v8f;

union Frag { v16h v; uint4 q[2]; };

// ---------------------------------------------------------------- fp32->fp16
__global__ void f32_to_f16(const float* __restrict__ src, _Float16* __restrict__ dst, int n) {
    int i = blockIdx.x * 256 + threadIdx.x;
    if (i < n) dst[i] = (_Float16)src[i];
}

// ------------------------------------------------- embedding + positional enc
__global__ void embed_pos(const float* __restrict__ embed, const int* __restrict__ tgt_pad,
                          const int* __restrict__ sos, float* __restrict__ x,
                          _Float16* __restrict__ xh) {
    int tn = blockIdx.x;           // t * NB_ + n
    int t = tn / NB_, n = tn % NB_;
    int idx = (t == 0) ? sos[0] : tgt_pad[n * TO_ + (t - 1)];
    const float sc = 22.627416998f;     // sqrt(512)
    for (int d = threadIdx.x; d < D_; d += 256) {
        int j = d >> 1;
        float freq = expf(-(float)(2 * j) * (9.2103403720f / 512.0f)); // ln(10000)/D
        float ang  = (float)t * freq;
        float pe   = (d & 1) ? cosf(ang) : sinf(ang);
        float v = embed[(size_t)idx * D_ + d] * sc + pe;
        x [(size_t)tn * D_ + d] = v;
        xh[(size_t)tn * D_ + d] = (_Float16)v;
    }
}

// ------------------------------------------------------------- WMMA GEMM core
// C[M x E] = A[M x K] (f16, row-major) @ W[E x K]^T (f16, row-major) + bias
// 256 threads = 8 waves; each wave owns TWO adjacent 16-wide E tiles
// (dual accumulators -> independent WMMA chains, A fragment reused 2x).
// Block computes a 16 x 256 output tile.
// If Cj != nullptr: joiner mode -> row m = (t*TI + i)*N + n is written to
// Cj[((n*TI + i)*T + t)*V + e] (the N x TI x T x V output layout).
__global__ void __launch_bounds__(256)
wmma_gemm(const _Float16* __restrict__ A, const _Float16* __restrict__ W,
          const float* __restrict__ bias, float* __restrict__ C,
          _Float16* __restrict__ Ch, float* __restrict__ Cj,
          int M, int K, int E, int relu) {
    __shared__ _Float16 Alds[16 * LDA_];
    const int tid   = threadIdx.x;
    const int lane  = tid & 31;
    const int wave  = tid >> 5;
    const int mtile = blockIdx.x;
    const int et0   = (blockIdx.y * 8 + wave) * 2;  // first of two E tiles
    const int col   = lane & 15;    // A-row within tile / B-column / C-column
    const int khalf = lane >> 4;    // upper/lower K half selector
    const bool act0 = (et0 * 16)       < E;
    const bool act1 = ((et0 + 1) * 16) < E;

    // clamp B row indices so inactive subtiles still issue in-bounds loads
    int r0 = et0 * 16 + col;       if (r0 >= E) r0 = 0;
    int r1 = (et0 + 1) * 16 + col; if (r1 >= E) r1 = r0;
    const _Float16* w0 = W + (size_t)r0 * K + khalf * 8;
    const _Float16* w1 = W + (size_t)r1 * K + khalf * 8;

    v8f acc0 = {}, acc1 = {};
    for (int kb = 0; kb < K; kb += 512) {
        __syncthreads();
        // stage A tile (16 rows x 512 halves) into LDS, 8 halves per uint4
        for (int idx = tid; idx < 1024; idx += 256) {
            int row = idx >> 6, cq = idx & 63;
            int m = mtile * 16 + row;
            uint4 val = make_uint4(0u, 0u, 0u, 0u);
            if (m < M) val = *(const uint4*)(A + (size_t)m * K + kb + cq * 8);
            *(uint4*)(&Alds[row * LDA_ + cq * 8]) = val;
        }
        __syncthreads();
        if (act0) {
            if (kb + 512 < K) {
                __builtin_prefetch(w0 + kb + 512, 0, 0);
                __builtin_prefetch(w1 + kb + 512, 0, 0);
            }
            for (int kk = 0; kk < 512; kk += 32) {
                Frag a, b0, b1;
                // A frag: lane row = col; j=0..7 -> K=kk+8*khalf+j ; j=8..15 -> +16
                const _Float16* ap = &Alds[col * LDA_ + kk + khalf * 8];
                a.q[0] = *(const uint4*)(ap);
                a.q[1] = *(const uint4*)(ap + 16);
                // B frags: lane col; j=0..7 -> K=kk+8*khalf+j ; j=8..15 -> +16
                b0.q[0] = *(const uint4*)(w0 + kb + kk);
                b0.q[1] = *(const uint4*)(w0 + kb + kk + 16);
                b1.q[0] = *(const uint4*)(w1 + kb + kk);
                b1.q[1] = *(const uint4*)(w1 + kb + kk + 16);
                // two independent accumulation chains keep the XDL pipe busy
                acc0 = __builtin_amdgcn_wmma_f32_16x16x32_f16(
                    false, a.v, false, b0.v, (short)0, acc0, false, false);
                acc1 = __builtin_amdgcn_wmma_f32_16x16x32_f16(
                    false, a.v, false, b1.v, (short)0, acc1, false, false);
            }
        }
    }
    for (int s = 0; s < 2; ++s) {
        if (s == 0 ? !act0 : !act1) continue;
        v8f acc = s ? acc1 : acc0;
        int e = (et0 + s) * 16 + col;
        float bv = bias ? bias[e] : 0.0f;
        for (int r = 0; r < 8; ++r) {
            int row = mtile * 16 + r + khalf * 8;  // C layout: VGPR r -> M=r+8*(lane/16)
            if (row >= M) continue;
            float v = acc[r] + bv;
            if (relu) v = fmaxf(v, 0.0f);
            if (Cj) {
                int t = row / (TI_ * NB_);
                int rem = row % (TI_ * NB_);
                int i = rem / NB_, n = rem % NB_;
                Cj[(((size_t)n * TI_ + i) * TT_ + t) * VV_ + e] = v;
            } else {
                C[(size_t)row * E + e] = v;
                if (Ch) Ch[(size_t)row * E + e] = (_Float16)v;
            }
        }
    }
}

// -------------------------------------------------------------- attention
// grid = (N*H, T), block = 64 threads; scores/softmax/AV for one (n,h,q).
__global__ void attention(const float* __restrict__ qkv, const int* __restrict__ tgt_len,
                          float* __restrict__ ao, _Float16* __restrict__ aoh) {
    __shared__ float s[TT_];
    int nh = blockIdx.x, qi = blockIdx.y;
    int n = nh / NH_, h = nh % NH_;
    int tid = threadIdx.x;
    int len = tgt_len[n]; if (len < 1) len = 1;

    if (tid < TT_) {
        int k = tid;
        const float* qp = qkv + ((size_t)qi * NB_ + n) * (3 * D_) + h * HD_;
        const float* kp = qkv + ((size_t)k  * NB_ + n) * (3 * D_) + D_ + h * HD_;
        float acc = 0.f;
        for (int d = 0; d < HD_; ++d) acc += qp[d] * kp[d];
        acc *= 0.125f;                       // 1/sqrt(64)
        if (k > qi || k >= len) acc = -1e9f; // causal | key padding
        s[tid] = acc;
    }
    __syncthreads();
    if (tid == 0) {
        float m = s[0];
        for (int k = 1; k < TT_; ++k) m = fmaxf(m, s[k]);
        float sum = 0.f;
        for (int k = 0; k < TT_; ++k) { float e = expf(s[k] - m); s[k] = e; sum += e; }
        float inv = 1.0f / sum;
        for (int k = 0; k < TT_; ++k) s[k] *= inv;
    }
    __syncthreads();
    if (tid < HD_) {
        float o = 0.f;
        for (int k = 0; k < TT_; ++k)
            o += s[k] * qkv[((size_t)k * NB_ + n) * (3 * D_) + 2 * D_ + h * HD_ + tid];
        size_t off = ((size_t)qi * NB_ + n) * D_ + h * HD_ + tid;
        ao[off]  = o;
        aoh[off] = (_Float16)o;
    }
}

// ----------------------------------------------- residual add + LayerNorm
__global__ void add_ln(float* __restrict__ x, const float* __restrict__ addv,
                       const float* __restrict__ s, const float* __restrict__ b,
                       _Float16* __restrict__ xh) {
    __shared__ float red[256];
    int tok = blockIdx.x, tid = threadIdx.x;
    float* xp = x + (size_t)tok * D_;
    const float* ap = addv + (size_t)tok * D_;
    int d0 = tid, d1 = tid + 256;
    float v0 = xp[d0] + ap[d0];
    float v1 = xp[d1] + ap[d1];
    red[tid] = v0 + v1; __syncthreads();
    for (int st = 128; st > 0; st >>= 1) { if (tid < st) red[tid] += red[tid + st]; __syncthreads(); }
    float mean = red[0] * (1.0f / D_); __syncthreads();
    red[tid] = v0 * v0 + v1 * v1; __syncthreads();
    for (int st = 128; st > 0; st >>= 1) { if (tid < st) red[tid] += red[tid + st]; __syncthreads(); }
    float var  = red[0] * (1.0f / D_) - mean * mean;
    float rstd = rsqrtf(var + 1e-5f);
    float y0 = (v0 - mean) * rstd * s[d0] + b[d0];
    float y1 = (v1 - mean) * rstd * s[d1] + b[d1];
    xp[d0] = y0; xp[d1] = y1;
    xh[(size_t)tok * D_ + d0] = (_Float16)y0;
    xh[(size_t)tok * D_ + d1] = (_Float16)y1;
}

// ------------------------------------------ joiner: tanh(enc + dec) -> fp16
__global__ void joiner_add(const float* __restrict__ encp, const float* __restrict__ x,
                           _Float16* __restrict__ addh) {
    int m = blockIdx.x;                         // (t*TI + i)*N + n
    int t = m / (TI_ * NB_);
    int rem = m % (TI_ * NB_);
    int i = rem / NB_, n = rem % NB_;
    const float* ep = encp + ((size_t)i * NB_ + n) * D_;
    const float* xp = x    + ((size_t)t * NB_ + n) * D_;
    _Float16* op = addh + (size_t)m * D_;
    for (int d = threadIdx.x; d < D_; d += 256)
        op[d] = (_Float16)tanhf(ep[d] + xp[d]);
}

// ---------------------------------------------------------------------------
extern "C" void kernel_launch(void* const* d_in, const int* in_sizes, int n_in,
                              void* d_out, int out_size, void* d_ws, size_t ws_size,
                              hipStream_t stream) {
    const float* enc_out = (const float*)d_in[0];
    const float* embed   = (const float*)d_in[1];
    const float* enc_w   = (const float*)d_in[2];
    const float* enc_b   = (const float*)d_in[3];
    const float* qkv_w   = (const float*)d_in[4];
    const float* qkv_b   = (const float*)d_in[5];
    const float* o_w     = (const float*)d_in[6];
    const float* o_b     = (const float*)d_in[7];
    const float* ln1_s   = (const float*)d_in[8];
    const float* ln1_b   = (const float*)d_in[9];
    const float* ff1_w   = (const float*)d_in[10];
    const float* ff1_b   = (const float*)d_in[11];
    const float* ff2_w   = (const float*)d_in[12];
    const float* ff2_b   = (const float*)d_in[13];
    const float* ln2_s   = (const float*)d_in[14];
    const float* ln2_b   = (const float*)d_in[15];
    const float* out_w   = (const float*)d_in[16];
    const int*   tgt_pad = (const int*)d_in[17];
    const int*   tgt_len = (const int*)d_in[18];
    const int*   sos     = (const int*)d_in[19];

    char* ws = (char*)d_ws;
    size_t off = 0;
    auto alloc = [&](size_t bytes) -> char* {
        char* p = ws + off;
        off = (off + bytes + 255) & ~(size_t)255;
        return p;
    };

    const int M   = TT_ * NB_;       // 264 decoder tokens
    const int ME  = TI_ * NB_;       // 1024 encoder tokens
    const int MJ  = TT_ * TI_ * NB_; // 33792 joiner rows

    _Float16* wh_qkv = (_Float16*)alloc((size_t)LL_ * 3 * D_ * D_ * 2);
    _Float16* wh_o   = (_Float16*)alloc((size_t)LL_ * D_ * D_ * 2);
    _Float16* wh_ff1 = (_Float16*)alloc((size_t)LL_ * FF_ * D_ * 2);
    _Float16* wh_ff2 = (_Float16*)alloc((size_t)LL_ * D_ * FF_ * 2);
    _Float16* wh_enc = (_Float16*)alloc((size_t)D_ * D_ * 2);
    _Float16* wh_out = (_Float16*)alloc((size_t)VV_ * D_ * 2);
    _Float16* ench   = (_Float16*)alloc((size_t)ME * D_ * 2);
    float*    x      = (float*)   alloc((size_t)M * D_ * 4);
    _Float16* xh     = (_Float16*)alloc((size_t)M * D_ * 2);
    float*    qkvbuf = (float*)   alloc((size_t)M * 3 * D_ * 4);
    float*    ao     = (float*)   alloc((size_t)M * D_ * 4);
    _Float16* aoh    = (_Float16*)alloc((size_t)M * D_ * 2);
    float*    tmp    = (float*)   alloc((size_t)M * D_ * 4);
    float*    ff1o   = (float*)   alloc((size_t)M * FF_ * 4);
    _Float16* ff1h   = (_Float16*)alloc((size_t)M * FF_ * 2);
    float*    encp   = (float*)   alloc((size_t)ME * D_ * 4);
    _Float16* addh   = (_Float16*)alloc((size_t)MJ * D_ * 2);

    auto cvt = [&](const float* s_, _Float16* d_, int n_) {
        f32_to_f16<<<(n_ + 255) / 256, 256, 0, stream>>>(s_, d_, n_);
    };
    cvt(qkv_w, wh_qkv, LL_ * 3 * D_ * D_);
    cvt(o_w,   wh_o,   LL_ * D_ * D_);
    cvt(ff1_w, wh_ff1, LL_ * FF_ * D_);
    cvt(ff2_w, wh_ff2, LL_ * D_ * FF_);
    cvt(enc_w, wh_enc, D_ * D_);
    cvt(out_w, wh_out, VV_ * D_);
    cvt(enc_out, ench, ME * D_);

    embed_pos<<<M, 256, 0, stream>>>(embed, tgt_pad, sos, x, xh);

    const int MT = (M + 15) / 16;   // 17 row tiles
    // grid.y = ceil(Etiles / 16): each block covers 16 E-tiles (8 waves x 2)
    for (int l = 0; l < LL_; ++l) {
        // QKV projection: 264x512 @ (1536x512)^T   (96 E-tiles -> 6)
        wmma_gemm<<<dim3(MT, 6), 256, 0, stream>>>(
            xh, wh_qkv + (size_t)l * 3 * D_ * D_, qkv_b + (size_t)l * 3 * D_,
            qkvbuf, nullptr, nullptr, M, D_, 3 * D_, 0);
        // self-attention (causal + key-padding)
        attention<<<dim3(NB_ * NH_, TT_), 64, 0, stream>>>(qkvbuf, tgt_len, ao, aoh);
        // output projection (32 E-tiles -> 2)
        wmma_gemm<<<dim3(MT, 2), 256, 0, stream>>>(
            aoh, wh_o + (size_t)l * D_ * D_, o_b + (size_t)l * D_,
            tmp, nullptr, nullptr, M, D_, D_, 0);
        add_ln<<<M, 256, 0, stream>>>(x, tmp, ln1_s + (size_t)l * D_, ln1_b + (size_t)l * D_, xh);
        // FFN (128 E-tiles -> 8 ; then 32 -> 2)
        wmma_gemm<<<dim3(MT, 8), 256, 0, stream>>>(
            xh, wh_ff1 + (size_t)l * FF_ * D_, ff1_b + (size_t)l * FF_,
            ff1o, ff1h, nullptr, M, D_, FF_, 1);
        wmma_gemm<<<dim3(MT, 2), 256, 0, stream>>>(
            ff1h, wh_ff2 + (size_t)l * D_ * FF_, ff2_b + (size_t)l * D_,
            tmp, nullptr, nullptr, M, FF_, D_, 0);
        add_ln<<<M, 256, 0, stream>>>(x, tmp, ln2_s + (size_t)l * D_, ln2_b + (size_t)l * D_, xh);
    }

    // encoder projection: 1024x512 @ (512x512)^T  (32 E-tiles -> 2)
    wmma_gemm<<<dim3(ME / 16, 2), 256, 0, stream>>>(
        ench, wh_enc, enc_b, encp, nullptr, nullptr, ME, D_, D_, 0);

    // joiner tanh(enc + dec) -> fp16
    joiner_add<<<MJ, 256, 0, stream>>>(encp, x, addh);

    // vocab projection: 33792x512 @ (2000x512)^T, write directly to d_out
    // (125 E-tiles -> 8 blocks of 16 tile-slots, last 3 slots idle)
    wmma_gemm<<<dim3(MJ / 16, 8), 256, 0, stream>>>(
        addh, wh_out, nullptr, nullptr, nullptr, (float*)d_out, MJ, D_, VV_, 0);
}
